// MultiHeadAtten_45475113730544
// MI455X (gfx1250) — matrix operand order repaired and checked by reading, not compiled, not run on hardware
//
#include <hip/hip_runtime.h>
#include <hip/hip_bf16.h>

typedef __attribute__((ext_vector_type(16))) _Float16 v16h;
typedef __attribute__((ext_vector_type(8)))  _Float16 v8h;
typedef __attribute__((ext_vector_type(8)))  float    v8f;

#define BB  2
#define SS  2048
#define DDm 1024
#define HHn 16
#define DH  64

static constexpr size_t N_X   = (size_t)BB * SS * DDm;      // 4,194,304
static constexpr size_t N_W   = (size_t)HHn * DDm * DH;     // 1,048,576
static constexpr size_t N_WO  = (size_t)DDm * DDm;          // 1,048,576
static constexpr size_t N_QKV = (size_t)BB * HHn * SS * DH; // 4,194,304

// half-element offsets inside workspace
static constexpr size_t O_XH  = 0;
static constexpr size_t O_WQ  = O_XH + N_X;   // transposed (H, DH, D)
static constexpr size_t O_WK  = O_WQ + N_W;
static constexpr size_t O_WV  = O_WK + N_W;
static constexpr size_t O_WO  = O_WV + N_W;   // natural (D, D)
static constexpr size_t O_Q   = O_WO + N_WO;  // (B*H, S, DH)
static constexpr size_t O_K   = O_Q  + N_QKV; // (B*H, S, DH)
static constexpr size_t O_V   = O_K  + N_QKV; // TRANSPOSED (B*H, DH, S)
static constexpr size_t O_CTX = O_V  + N_QKV; // (B*S, D)

// ---------------- WMMA + fragment helpers (ISA 7.12.2 layouts, wave32) ------

__device__ __forceinline__ v8f wmma16x16x32(v16h a, v16h b, v8f c) {
  return __builtin_amdgcn_wmma_f32_16x16x32_f16(
      false, a, false, b, (short)0, c, false, false);
}

__device__ __forceinline__ v16h pack16(v8h a0, v8h a1) {
  v16h r;
#pragma unroll
  for (int i = 0; i < 8; ++i) { r[i] = a0[i]; r[i + 8] = a1[i]; }
  return r;
}

// A 16x32 (MxK), K contiguous in memory (row-major, stride strideM halves).
// lane l<16: row l, K = hi*16 .. hi*16+15; hi = lane>>4
__device__ __forceinline__ v16h load_a_contig(const _Float16* src, int strideM,
                                              int lane) {
  int m  = lane & 15;
  int hi = (lane >> 4) & 1;
  const _Float16* r0 = src + (size_t)m * strideM + hi * 16;
  v8h a0 = *(const v8h*)r0;         // K run 1 (16B)
  v8h a1 = *(const v8h*)(r0 + 8);   // K run 2 (16B)
  return pack16(a0, a1);
}

// B 32x16 (KxN), memory holds B^T: element(K,N) = src[n*strideN + k]
// (K contiguous per lane). lane l: col l&15, K = hi*16 .. hi*16+15.
__device__ __forceinline__ v16h load_b_contig(const _Float16* src,
                                              size_t strideN, int lane) {
  int n  = lane & 15;
  int hi = (lane >> 4) & 1;
  return *(const v16h*)(src + (size_t)n * strideN + hi * 16);  // 32B
}

// C/D 16x16 f32: lane l<16 -> col l, rows 0..7; lane l>=16 -> col l-16, rows 8..15
__device__ __forceinline__ float half_reduce_max(float v) {
  v = fmaxf(v, __shfl_xor(v, 1, 32));
  v = fmaxf(v, __shfl_xor(v, 2, 32));
  v = fmaxf(v, __shfl_xor(v, 4, 32));
  v = fmaxf(v, __shfl_xor(v, 8, 32));
  return v;
}
__device__ __forceinline__ float half_reduce_sum(float v) {
  v += __shfl_xor(v, 1, 32);
  v += __shfl_xor(v, 2, 32);
  v += __shfl_xor(v, 4, 32);
  v += __shfl_xor(v, 8, 32);
  return v;
}
__device__ __forceinline__ void lds_fence() {
  asm volatile("s_wait_dscnt 0" ::: "memory");
}
__device__ __forceinline__ void async_wait0() {
  asm volatile("s_wait_asynccnt 0x0" ::: "memory");
}
// CDNA5 async DMA: global -> LDS, 16 bytes per lane, tracked by ASYNCcnt
__device__ __forceinline__ void async_g2lds_b128(const _Float16* gaddr,
                                                 _Float16* lds_ptr) {
  unsigned laddr = (unsigned)(size_t)lds_ptr;  // low 32 bits = LDS offset
  asm volatile("global_load_async_to_lds_b128 %0, %1, off"
               :: "v"(laddr), "v"(gaddr) : "memory");
}

// ---------------- Kernel 0a: f32 -> f16 convert ------------------------------

__global__ void cvt_f32_f16(const float* __restrict__ src,
                            _Float16* __restrict__ dst, int n) {
  int i = blockIdx.x * blockDim.x + threadIdx.x;
  int stride = gridDim.x * blockDim.x;
  for (; i < n; i += stride) dst[i] = (_Float16)src[i];
}

// ---------------- Kernel 0b: weight transpose-convert (H,D,DH)->(H,DH,D) -----

__global__ void cvt_w_transpose(const float* __restrict__ src,
                                _Float16* __restrict__ dst) {
  size_t i = (size_t)blockIdx.x * blockDim.x + threadIdx.x;
  if (i >= N_W) return;
  size_t h   = i / ((size_t)DDm * DH);
  size_t rem = i % ((size_t)DDm * DH);
  size_t d = rem / DH, e = rem % DH;
  dst[h * DH * DDm + e * DDm + d] = (_Float16)src[i];
}

// ---------------- Kernel 1: QKV projection -----------------------------------
// grid (S/64, B*H, 3), block 128. Wave: 16 rows x 64 cols.
// q,k written natural (S,DH); v written transposed (DH,S).
__global__ void qkv_proj(const _Float16* __restrict__ xh,
                         const _Float16* __restrict__ wqT,
                         const _Float16* __restrict__ wkT,
                         const _Float16* __restrict__ wvT,
                         _Float16* __restrict__ qb,
                         _Float16* __restrict__ kb,
                         _Float16* __restrict__ vT) {
  int lane = threadIdx.x & 31;
  int wave = threadIdx.x >> 5;
  int bh = blockIdx.y;
  int b  = bh / HHn;
  int h  = bh % HHn;
  int s0 = blockIdx.x * 64 + wave * 16;
  int which = blockIdx.z;

  const _Float16* WT = (which == 0 ? wqT : which == 1 ? wkT : wvT) +
                       (size_t)h * DH * DDm;  // (DH, D)
  const _Float16* X = xh + ((size_t)b * SS + s0) * DDm;

  v8f zero = {};
  v8f acc[4] = {zero, zero, zero, zero};

  for (int kk = 0; kk < DDm; kk += 32) {
    v16h a = load_a_contig(X + kk, DDm, lane);
#pragma unroll
    for (int t = 0; t < 4; ++t) {
      // B(K=d, N=e) = WT[e][d] : per-lane contiguous in d
      v16h bf = load_b_contig(WT + (size_t)(t * 16) * DDm + kk, DDm, lane);
      acc[t] = wmma16x16x32(a, bf, acc[t]);
    }
  }

  int n  = lane & 15;
  int hi = (lane >> 4) & 1;
  if (which == 2) {
    // vT[(bh*DH + e)*S + s] ; rows hi*8+r are consecutive s -> one b128 store
#pragma unroll
    for (int t = 0; t < 4; ++t) {
      v8h h8;
#pragma unroll
      for (int r = 0; r < 8; ++r) h8[r] = (_Float16)acc[t][r];
      *(v8h*)(vT + ((size_t)bh * DH + t * 16 + n) * SS + s0 + hi * 8) = h8;
    }
  } else {
    _Float16* out = (which == 0 ? qb : kb);
#pragma unroll
    for (int t = 0; t < 4; ++t)
#pragma unroll
      for (int r = 0; r < 8; ++r) {
        int row = hi * 8 + r;
        out[((size_t)bh * SS + s0 + row) * DH + t * 16 + n] =
            (_Float16)acc[t][r];
      }
  }
}

// ---------------- Kernel 2: fused flash attention ----------------------------
// grid (S/64, B*H), block 128 (4 waves, one 16-query tile each).
// K tile (32 keys x 64) staged to LDS via async DMA, shared by the 4 waves.
__global__ void flash_attn(const _Float16* __restrict__ qb,
                           const _Float16* __restrict__ kbuf,
                           const _Float16* __restrict__ vT,
                           const unsigned char* __restrict__ mask,
                           _Float16* __restrict__ ctxh) {
  __shared__ _Float16 Ktile[32 * 64];       // [key][e]
  __shared__ _Float16 plds[4][16 * 32];     // per-wave P tile

  int lane = threadIdx.x & 31;
  int wave = threadIdx.x >> 5;
  int tid  = threadIdx.x;
  int bh = blockIdx.y;
  int b  = bh / HHn;
  int h  = bh % HHn;
  int q0 = blockIdx.x * 64 + wave * 16;
  int n  = lane & 15;
  int hi = (lane >> 4) & 1;
  const float scale = 0.125f;  // 1/sqrt(64)

  const _Float16* qp = qb + ((size_t)bh * SS + q0) * DH;
  v16h qa0 = load_a_contig(qp,      DH, lane);  // e 0..31
  v16h qa1 = load_a_contig(qp + 32, DH, lane);  // e 32..63

  const _Float16* kbase = kbuf + (size_t)bh * SS * DH;
  const _Float16* vbase = vT + (size_t)bh * DH * SS;

  v8f zero = {};
  v8f acc[4] = {zero, zero, zero, zero};
  float m_r[8], l_r[8];
#pragma unroll
  for (int r = 0; r < 8; ++r) { m_r[r] = -3.0e38f; l_r[r] = 0.0f; }

  _Float16* P = plds[wave];

  for (int kb = 0; kb < SS; kb += 32) {
    // --- stage K tile (32x64 halves, contiguous 4KB in global) via async DMA
    const _Float16* kp = kbase + (size_t)kb * DH;
#pragma unroll
    for (int c = 0; c < 2; ++c) {
      int off = (c * 128 + tid) * 8;  // halves
      async_g2lds_b128(kp + off, Ktile + off);
    }
    async_wait0();
    __syncthreads();

    // --- scores: 16 queries x 32 keys, from LDS K tile
    v8f s0 = zero, s1 = zero;
    s0 = wmma16x16x32(qa0, load_b_contig(Ktile,                 64, lane), s0);
    s0 = wmma16x16x32(qa1, load_b_contig(Ktile + 32,            64, lane), s0);
    s1 = wmma16x16x32(qa0, load_b_contig(Ktile + 16 * 64,       64, lane), s1);
    s1 = wmma16x16x32(qa1, load_b_contig(Ktile + 16 * 64 + 32,  64, lane), s1);

    // key mask + scale
    float km0 = mask[(size_t)b * SS + kb + n]      ? 0.0f : -1.0e10f;
    float km1 = mask[(size_t)b * SS + kb + 16 + n] ? 0.0f : -1.0e10f;
#pragma unroll
    for (int r = 0; r < 8; ++r) {
      s0[r] = s0[r] * scale + km0;
      s1[r] = s1[r] * scale + km1;
    }

    // online softmax
    float p0[8], p1[8];
#pragma unroll
    for (int r = 0; r < 8; ++r) {
      float rm = half_reduce_max(fmaxf(s0[r], s1[r]));
      float mnew = fmaxf(m_r[r], rm);
      float alpha = __expf(m_r[r] - mnew);
      m_r[r] = mnew;
      p0[r] = __expf(s0[r] - mnew);
      p1[r] = __expf(s1[r] - mnew);
      float rs = half_reduce_sum(p0[r] + p1[r]);
      l_r[r] = l_r[r] * alpha + rs;
#pragma unroll
      for (int t = 0; t < 4; ++t) acc[t][r] *= alpha;
    }

    // stage P (16x32) in per-wave LDS, reload as A fragment
#pragma unroll
    for (int r = 0; r < 8; ++r) {
      P[(hi * 8 + r) * 32 + n]      = (_Float16)p0[r];
      P[(hi * 8 + r) * 32 + 16 + n] = (_Float16)p1[r];
    }
    lds_fence();
    v16h pa = load_a_contig(P, 32, lane);

    // ctx += P @ V ; B(K=key, N=e) = vT[e][key] -> per-lane contiguous keys
    const _Float16* vp = vbase + kb;
#pragma unroll
    for (int t = 0; t < 4; ++t) {
      v16h bf = load_b_contig(vp + (size_t)(t * 16) * SS, SS, lane);
      acc[t] = wmma16x16x32(pa, bf, acc[t]);
    }

    __syncthreads();  // protect Ktile before next iteration's DMA
  }

  // finalize: /l, query mask, store ctx (b, s, h*DH+e) f16
#pragma unroll
  for (int r = 0; r < 8; ++r) {
    int row = hi * 8 + r;
    float qm = mask[(size_t)b * SS + q0 + row] ? 1.0f : 0.0f;
    float inv = qm / l_r[r];
#pragma unroll
    for (int t = 0; t < 4; ++t) {
      ctxh[((size_t)b * SS + q0 + row) * DDm + h * DH + t * 16 + n] =
          (_Float16)(acc[t][r] * inv);
    }
  }
}

// ---------------- Kernel 3: output projection (ctx @ Wo^T, f32 out) ----------
// grid ((B*S)/64, D/64), block 128.
__global__ void out_proj(const _Float16* __restrict__ ctxh,
                         const _Float16* __restrict__ woh,
                         float* __restrict__ out) {
  int lane = threadIdx.x & 31;
  int wave = threadIdx.x >> 5;
  int row0 = blockIdx.x * 64 + wave * 16;  // over B*S
  int col0 = blockIdx.y * 64;              // over D

  const _Float16* A = ctxh + (size_t)row0 * DDm;

  v8f zero = {};
  v8f acc[4] = {zero, zero, zero, zero};

  for (int kk = 0; kk < DDm; kk += 32) {
    v16h a = load_a_contig(A + kk, DDm, lane);
#pragma unroll
    for (int t = 0; t < 4; ++t) {
      // B(K=e, N=d) = Wo[d][e] : per-lane contiguous in e
      v16h bf = load_b_contig(woh + (size_t)(col0 + t * 16) * DDm + kk,
                              DDm, lane);
      acc[t] = wmma16x16x32(a, bf, acc[t]);
    }
  }

  int n  = lane & 15;
  int hi = (lane >> 4) & 1;
#pragma unroll
  for (int t = 0; t < 4; ++t)
#pragma unroll
    for (int r = 0; r < 8; ++r) {
      int row = hi * 8 + r;
      out[(size_t)(row0 + row) * DDm + col0 + t * 16 + n] = acc[t][r];
    }
}

// ---------------- launch -----------------------------------------------------

extern "C" void kernel_launch(void* const* d_in, const int* in_sizes, int n_in,
                              void* d_out, int out_size, void* d_ws, size_t ws_size,
                              hipStream_t stream) {
  const float* x   = (const float*)d_in[0];
  const unsigned char* mask = (const unsigned char*)d_in[1];  // jax bool, 1B
  const float* Wq  = (const float*)d_in[2];
  const float* Wk  = (const float*)d_in[3];
  const float* Wv  = (const float*)d_in[4];
  const float* Wo  = (const float*)d_in[5];
  float* out = (float*)d_out;

  _Float16* ws = (_Float16*)d_ws;
  _Float16* xh   = ws + O_XH;
  _Float16* wqT  = ws + O_WQ;
  _Float16* wkT  = ws + O_WK;
  _Float16* wvT  = ws + O_WV;
  _Float16* woh  = ws + O_WO;
  _Float16* qb   = ws + O_Q;
  _Float16* kb   = ws + O_K;
  _Float16* vT   = ws + O_V;
  _Float16* ctxh = ws + O_CTX;

  // phase 0: convert (x, Wo) and transpose-convert (Wq, Wk, Wv)
  cvt_f32_f16<<<1024, 256, 0, stream>>>(x,  xh,  (int)N_X);
  cvt_f32_f16<<<512,  256, 0, stream>>>(Wo, woh, (int)N_WO);
  cvt_w_transpose<<<(int)(N_W / 256), 256, 0, stream>>>(Wq, wqT);
  cvt_w_transpose<<<(int)(N_W / 256), 256, 0, stream>>>(Wk, wkT);
  cvt_w_transpose<<<(int)(N_W / 256), 256, 0, stream>>>(Wv, wvT);

  // phase 1: QKV projections (v stored transposed)
  dim3 g1(SS / 64, BB * HHn, 3);
  qkv_proj<<<g1, 128, 0, stream>>>(xh, wqT, wkT, wvT, qb, kb, vT);

  // phase 2: fused flash attention (async K-tile staging)
  dim3 g2(SS / 64, BB * HHn);
  flash_attn<<<g2, 128, 0, stream>>>(qb, kb, vT, mask, ctxh);

  // phase 3: output projection
  dim3 g3((BB * SS) / 64, DDm / 64);
  out_proj<<<g3, 128, 0, stream>>>(ctxh, woh, out);
}
